// BahdanauAttention_4647154614502
// MI455X (gfx1250) — matrix operand order repaired and checked by reading, not compiled, not run on hardware
//
#include <hip/hip_runtime.h>
#include <stdint.h>

#define B_   32
#define T_   512
#define DH_  2048
#define MTOT (B_ * T_)   // 16384

#define BM 128
#define BN 128
#define BK 32
#define LDA 40           // padded LDS row length in bf16 elems (80B: 16B aligned, good bank spread)

typedef __attribute__((ext_vector_type(16))) __bf16 v16bf;
typedef __attribute__((ext_vector_type(8)))  float  v8f;

union Frag { v16bf v; uint4 q[2]; };

__device__ __forceinline__ unsigned short f2bf(float x) {
  unsigned u = __float_as_uint(x);
  u += 0x7FFFu + ((u >> 16) & 1u);   // round-to-nearest-even
  return (unsigned short)(u >> 16);
}

// async DMA: 16B global -> LDS, tracked by ASYNCcnt (CDNA5 GLOBAL_LOAD_ASYNC_TO_LDS_B128)
__device__ __forceinline__ void async_b128(const __bf16* g, unsigned lds_byte_off) {
  asm volatile("global_load_async_to_lds_b128 %0, %1, off"
               :: "v"(lds_byte_off), "v"(g) : "memory");
}
__device__ __forceinline__ void wait_async0() {
  asm volatile("s_wait_asynccnt 0x0" ::: "memory");
}

// ---- pack f32 -> bf16 (2 at a time) ----------------------------------------
__global__ void cvt_pk_bf16(const float2* __restrict__ src, uint32_t* __restrict__ dst, int n2) {
  int i = blockIdx.x * blockDim.x + threadIdx.x;
  if (i >= n2) return;
  float2 v = src[i];
  dst[i] = (uint32_t)f2bf(v.x) | ((uint32_t)f2bf(v.y) << 16);
}

// ---- small [32,2048] = x @ W^T + b (optionally pass-through when *t != 0) --
__global__ void linear_bdh(const float* __restrict__ x, const float* __restrict__ W,
                           const float* __restrict__ bias, float* __restrict__ out,
                           const int* __restrict__ tflag) {
  int idx = blockIdx.x * blockDim.x + threadIdx.x;   // B_*DH_ threads
  int row = idx >> 11, col = idx & (DH_ - 1);
  if (tflag && *tflag != 0) { out[idx] = x[idx]; return; }
  const float4* xr = (const float4*)(x + (size_t)row * DH_);
  const float4* wr = (const float4*)(W + (size_t)col * DH_);
  float acc = bias[col];
#pragma unroll 4
  for (int i = 0; i < DH_ / 4; ++i) {
    float4 a = xr[i], w = wr[i];
    acc = fmaf(a.x, w.x, acc); acc = fmaf(a.y, w.y, acc);
    acc = fmaf(a.z, w.z, acc); acc = fmaf(a.w, w.w, acc);
  }
  out[idx] = acc;
}

__global__ void init_scores(float* __restrict__ scores, const float* __restrict__ vab) {
  int i = blockIdx.x * blockDim.x + threadIdx.x;
  if (i < MTOT) scores[i] = vab[0];
}

// ---- fused: scores[m] += sum_n Va[n] * tanh(q[b][n] + Ua_b[n] + (keys @ Ua^T)[m][n])
__global__ __launch_bounds__(256)
void fused_score(const __bf16* __restrict__ Kb,   // [MTOT, DH] bf16
                 const __bf16* __restrict__ Ub,   // [DH, DH]  bf16 (Ua_w, row e holds k-contig)
                 const float*  __restrict__ qv,   // [B_, DH]
                 const float*  __restrict__ uab,  // [DH]
                 const float*  __restrict__ va,   // [DH]
                 float*        __restrict__ scores) {
  __shared__ __bf16 As[2][BM][LDA];   // double-buffered
  __shared__ __bf16 Bs[2][BN][LDA];
  __shared__ float  sc[BM];

  const int tid  = threadIdx.x;
  const int lane = tid & 31;
  const int wave = tid >> 5;
  const int h    = (lane >> 4) & 1;    // K-half selector per ISA layout
  const int ln   = lane & 15;
  const int wm   = wave >> 1;          // 0..3  -> 32 M-rows per wave
  const int wn   = wave & 1;           // 0..1  -> 64 N-cols per wave

  const int m0 = blockIdx.x * BM;
  const int n0 = blockIdx.y * BN;
  const int b  = m0 >> 9;              // T_=512; all 128 rows share one batch index

  if (tid < BM) sc[tid] = 0.0f;

  // staging: each thread DMA-copies 32B of A and 32B of B per K-step
  const int lrow  = tid >> 1;          // 0..127
  const int lhalf = (tid & 1) * 16;    // bf16-element offset 0 or 16
  const __bf16* gA = Kb + (size_t)(m0 + lrow) * DH_ + lhalf;
  const __bf16* gB = Ub + (size_t)(n0 + lrow) * DH_ + lhalf;

  // prologue: DMA first K-tile into buffer 0
  async_b128(gA,     (unsigned)(uintptr_t)&As[0][lrow][lhalf]);
  async_b128(gA + 8, (unsigned)(uintptr_t)&As[0][lrow][lhalf + 8]);
  async_b128(gB,     (unsigned)(uintptr_t)&Bs[0][lrow][lhalf]);
  async_b128(gB + 8, (unsigned)(uintptr_t)&Bs[0][lrow][lhalf + 8]);

  v8f acc[2][4] = {};
  int buf = 0;

  for (int kk = 0; kk < DH_; kk += BK) {
    wait_async0();        // own DMA into 'buf' complete
    __syncthreads();      // everyone's DMA complete; prev readers of buf^1 done

    // load all fragments for this K-step up front, then issue WMMAs back-to-back
    Frag af[2];
#pragma unroll
    for (int i = 0; i < 2; ++i) {
      const int m = wm * 32 + i * 16 + ln;
      af[i].q[0] = *(const uint4*)&As[buf][m][h * 8];        // 16-bit A 16x32 layout
      af[i].q[1] = *(const uint4*)&As[buf][m][16 + h * 8];
    }
    Frag bfr[4];
#pragma unroll
    for (int j = 0; j < 4; ++j) {
      const int n = wn * 64 + j * 16 + ln;
      bfr[j].q[0] = *(const uint4*)&Bs[buf][n][h * 8];      // 16-bit B 32x16 layout
      bfr[j].q[1] = *(const uint4*)&Bs[buf][n][16 + h * 8];
    }

    // overlap: DMA next K-tile into the other buffer while WMMAs run
    const int kn = kk + BK;
    if (kn < DH_) {
      const int nb = buf ^ 1;
      async_b128(gA + kn,     (unsigned)(uintptr_t)&As[nb][lrow][lhalf]);
      async_b128(gA + kn + 8, (unsigned)(uintptr_t)&As[nb][lrow][lhalf + 8]);
      async_b128(gB + kn,     (unsigned)(uintptr_t)&Bs[nb][lrow][lhalf]);
      async_b128(gB + kn + 8, (unsigned)(uintptr_t)&Bs[nb][lrow][lhalf + 8]);
    }

#pragma unroll
    for (int j = 0; j < 4; ++j) {
#pragma unroll
      for (int i = 0; i < 2; ++i) {
        acc[i][j] = __builtin_amdgcn_wmma_f32_16x16x32_bf16(
            false, af[i].v, false, bfr[j].v, (short)0, acc[i][j], false, false);
      }
    }
    buf ^= 1;
  }

  // epilogue: tanh + Va-weighted reduction over this block's N columns
  float qvl[4], ubl[4], val[4];
#pragma unroll
  for (int j = 0; j < 4; ++j) {
    const int n = n0 + wn * 64 + j * 16 + ln;
    qvl[j] = qv[(size_t)b * DH_ + n];
    ubl[j] = uab[n];
    val[j] = va[n];
  }
#pragma unroll
  for (int i = 0; i < 2; ++i) {
#pragma unroll
    for (int r = 0; r < 8; ++r) {
      float s = 0.0f;
#pragma unroll
      for (int j = 0; j < 4; ++j) {
        float e = tanhf(qvl[j] + ubl[j] + acc[i][j][r]);
        s = fmaf(val[j], e, s);
      }
      atomicAdd(&sc[wm * 32 + i * 16 + 8 * h + r], s);   // ds_add_f32
    }
  }
  __syncthreads();
  if (tid < BM) atomicAdd(&scores[m0 + tid], sc[tid]);   // cross-N-block accumulation
}

// ---- softmax over T per batch row ------------------------------------------
__global__ void softmax_rows(const float* __restrict__ scores, float* __restrict__ attn) {
  __shared__ float red[256];
  const int b = blockIdx.x, tid = threadIdx.x;
  float s0 = scores[b * T_ + tid];
  float s1 = scores[b * T_ + tid + 256];
  red[tid] = fmaxf(s0, s1);
  __syncthreads();
  for (int off = 128; off > 0; off >>= 1) {
    if (tid < off) red[tid] = fmaxf(red[tid], red[tid + off]);
    __syncthreads();
  }
  const float mx = red[0];
  __syncthreads();
  float e0 = __expf(s0 - mx), e1 = __expf(s1 - mx);
  red[tid] = e0 + e1;
  __syncthreads();
  for (int off = 128; off > 0; off >>= 1) {
    if (tid < off) red[tid] += red[tid + off];
    __syncthreads();
  }
  const float inv = 1.0f / red[0];
  attn[b * T_ + tid]       = e0 * inv;
  attn[b * T_ + tid + 256] = e1 * inv;
}

// ---- attn_values[b][d] = sum_t attn[b][t] * keys[b][t][d] (f32 keys) -------
__global__ void weighted_sum(const float* __restrict__ attn, const float* __restrict__ keys,
                             float* __restrict__ out) {
  const int idx = blockIdx.x * blockDim.x + threadIdx.x;  // B_*DH_
  const int b = idx >> 11, d = idx & (DH_ - 1);
  const float* kb = keys + (size_t)b * T_ * DH_ + d;
  const float* ab = attn + b * T_;
  float acc = 0.0f;
#pragma unroll 8
  for (int t = 0; t < T_; ++t) acc = fmaf(ab[t], kb[(size_t)t * DH_], acc);
  out[idx] = acc;
}

extern "C" void kernel_launch(void* const* d_in, const int* in_sizes, int n_in,
                              void* d_out, int out_size, void* d_ws, size_t ws_size,
                              hipStream_t stream) {
  const float* keys = (const float*)d_in[0];
  const float* dec  = (const float*)d_in[1];
  const float* Wsw  = (const float*)d_in[2];
  const float* Wsb  = (const float*)d_in[3];
  const float* Waw  = (const float*)d_in[4];
  const float* Wab  = (const float*)d_in[5];
  const float* Uaw  = (const float*)d_in[6];
  const float* Uab  = (const float*)d_in[7];
  const float* Vaw  = (const float*)d_in[8];
  const float* Vab  = (const float*)d_in[9];
  const int*   tpt  = (const int*)d_in[10];

  char* ws = (char*)d_ws;
  __bf16* keys_bf = (__bf16*)ws;                                        // 64 MB
  __bf16* ua_bf   = (__bf16*)(ws + (size_t)MTOT * DH_ * 2);             //  8 MB
  float*  query   = (float*)(ws + (size_t)MTOT * DH_ * 2 + (size_t)DH_ * DH_ * 2);
  float*  qbuf    = query + B_ * DH_;
  float*  scores  = qbuf + B_ * DH_;

  float* out_vals = (float*)d_out;            // [B, DH]
  float* out_attn = out_vals + B_ * DH_;      // [B, T, 1]

  // 1) pack keys and Ua_w to bf16 for WMMA
  {
    const int n2 = MTOT * DH_ / 2;
    cvt_pk_bf16<<<n2 / 256, 256, 0, stream>>>((const float2*)keys, (uint32_t*)keys_bf, n2);
    const int w2 = DH_ * DH_ / 2;
    cvt_pk_bf16<<<w2 / 256, 256, 0, stream>>>((const float2*)Uaw, (uint32_t*)ua_bf, w2);
  }
  // 2) query = (t==0 ? dec @ Ws^T + Ws_b : dec); q = query @ Wa^T + Wa_b
  linear_bdh<<<B_ * DH_ / 256, 256, 0, stream>>>(dec,   Wsw, Wsb, query, tpt);
  linear_bdh<<<B_ * DH_ / 256, 256, 0, stream>>>(query, Waw, Wab, qbuf,  nullptr);
  // 3) scores = Va_b
  init_scores<<<MTOT / 256, 256, 0, stream>>>(scores, Vab);
  // 4) fused bf16-WMMA GEMM + tanh + Va reduction (async-DMA double-buffered)
  {
    dim3 grid(MTOT / BM, DH_ / BN);
    fused_score<<<grid, 256, 0, stream>>>(keys_bf, ua_bf, qbuf, Uab, Vaw, scores);
  }
  // 5) softmax over T
  softmax_rows<<<B_, 256, 0, stream>>>(scores, out_attn);
  // 6) attn-weighted sum of original f32 keys
  weighted_sum<<<B_ * DH_ / 256, 256, 0, stream>>>(out_attn, keys, out_vals);
}
